// MultiheadAttentionBlock_15436112462370
// MI455X (gfx1250) — matrix operand order repaired and checked by reading, not compiled, not run on hardware
//
#include <hip/hip_runtime.h>
#include <stdint.h>

typedef __attribute__((ext_vector_type(16))) __bf16        v16bf;
typedef __attribute__((ext_vector_type(8)))  float         v8f;
typedef __attribute__((ext_vector_type(4)))  unsigned int  u32x4;
typedef __attribute__((ext_vector_type(2)))  unsigned int  u32x2;
typedef __attribute__((ext_vector_type(4)))  float         f32x4;
typedef __attribute__((ext_vector_type(4)))  int           i32x4;

#define GLOBAL_AS __attribute__((address_space(1)))
#define LDS_AS    __attribute__((address_space(3)))

#define D_MODEL   1024
#define NUM_HEADS 16
#define HEAD_DIM  64
#define T_SEQ     2048
#define B_BATCH   2
#define M_ROWS    (B_BATCH * T_SEQ)   // 4096

__device__ __forceinline__ unsigned short f32_to_bf16(float f) {
  union { float f; uint32_t u; } c;
  c.f = f;
  uint32_t u = c.u;
  u += 0x7FFFu + ((u >> 16) & 1u);    // round-to-nearest-even
  return (unsigned short)(u >> 16);
}

__device__ __forceinline__ unsigned int pack_bf16x2(float lo, float hi) {
  return (unsigned int)f32_to_bf16(lo) | ((unsigned int)f32_to_bf16(hi) << 16);
}

union Frag { u32x4 u[2]; v16bf v; };  // 32 bytes: one 16-bit WMMA A/B fragment

// ---------------------------------------------------------------------------
// Generic 4096x1024x1024 GEMM:  C = A @ W + bias   (double-buffered LDS)
//   A: f32 [4096][1024] (A32) or bf16 [4096][1024] (A16)
//   W: f32 [1024][1024] row-major ([in][out], matching x @ W)
// mode 0: bf16 out at [((b*H+h)*T + t)*64 + hd]      (Q / K per-head layout)
// mode 2: bf16 out at [((b*H+h)*64 + hd)*T + t]      (V transposed per head)
// mode 3: f32  out at [m*1024 + n]                   (final output)
// ---------------------------------------------------------------------------
__global__ __launch_bounds__(256) void gemm_bf16wmma(
    const float* __restrict__ A32, const unsigned short* __restrict__ A16,
    const float* __restrict__ W, const float* __restrict__ bias,
    unsigned short* __restrict__ out16, float* __restrict__ out32, int mode)
{
  __shared__ unsigned short As[2][128 * 40];  // [row][k], stride 40 halves (80B)
  __shared__ unsigned short Bs[2][128 * 40];  // [n][k]  (W tile transposed)

  const int tid  = threadIdx.x;
  const int wave = tid >> 5;
  const int lane = tid & 31;
  const int half = lane >> 4;
  const int l16  = lane & 15;
  const int wm   = wave & 3;                  // wave M group (0..3)
  const int wn   = wave >> 2;                 // wave N group (0..1)
  const int mBlock = blockIdx.y * 128;
  const int nBlock = blockIdx.x * 128;

  v8f acc[2][4];
  #pragma unroll
  for (int mi = 0; mi < 2; ++mi)
    #pragma unroll
    for (int nj = 0; nj < 4; ++nj)
      #pragma unroll
      for (int i = 0; i < 8; ++i) acc[mi][nj][i] = 0.0f;

  f32x4 aR[4];                                // staged A (f32 path)
  u32x4 aR16[2];                              // staged A (bf16 path)
  f32x4 wR[4];                                // staged W

  auto loadTiles = [&](int kt) {
    if (A32) {
      #pragma unroll
      for (int i = 0; i < 4; ++i) {
        int id = tid + i * 256;               // 0..1023
        int r  = id >> 3;                     // 0..127
        int c  = (id & 7) * 4;                // 0,4,..,28
        aR[i] = *(const f32x4*)(A32 + (size_t)(mBlock + r) * D_MODEL + kt + c);
      }
    } else {
      #pragma unroll
      for (int i = 0; i < 2; ++i) {
        int id = tid + i * 256;               // 0..511
        int r  = id >> 2;                     // 0..127
        int c  = (id & 3) * 8;                // 0,8,16,24
        aR16[i] = *(const u32x4*)(A16 + (size_t)(mBlock + r) * D_MODEL + kt + c);
      }
    }
    #pragma unroll
    for (int i = 0; i < 4; ++i) {
      int id = tid + i * 256;                 // 0..1023
      int kr = id >> 5;                       // 0..31
      int nc = (id & 31) * 4;                 // 0..124
      wR[i] = *(const f32x4*)(W + (size_t)(kt + kr) * D_MODEL + nBlock + nc);
    }
  };

  auto storeTiles = [&](int buf) {
    if (A32) {
      #pragma unroll
      for (int i = 0; i < 4; ++i) {
        int id = tid + i * 256;
        int r  = id >> 3;
        int c  = (id & 7) * 4;
        u32x2 p = { pack_bf16x2(aR[i].x, aR[i].y), pack_bf16x2(aR[i].z, aR[i].w) };
        *(u32x2*)&As[buf][r * 40 + c] = p;    // 8B-aligned ds_store_b64
      }
    } else {
      #pragma unroll
      for (int i = 0; i < 2; ++i) {
        int id = tid + i * 256;
        int r  = id >> 2;
        int c  = (id & 3) * 8;
        *(u32x4*)&As[buf][r * 40 + c] = aR16[i];
      }
    }
    #pragma unroll
    for (int i = 0; i < 4; ++i) {             // transpose scatter into Bs[n][k]
      int id = tid + i * 256;
      int kr = id >> 5;
      int nc = (id & 31) * 4;
      Bs[buf][(nc + 0) * 40 + kr] = f32_to_bf16(wR[i].x);
      Bs[buf][(nc + 1) * 40 + kr] = f32_to_bf16(wR[i].y);
      Bs[buf][(nc + 2) * 40 + kr] = f32_to_bf16(wR[i].z);
      Bs[buf][(nc + 3) * 40 + kr] = f32_to_bf16(wR[i].w);
    }
  };

  loadTiles(0);
  storeTiles(0);
  __syncthreads();

  for (int it = 0; it < 32; ++it) {
    const int cur = it & 1;
    if (it + 1 < 32) loadTiles((it + 1) * 32);  // global loads overlap WMMA below

    // A fragment: lane row = m; halves j<8 -> k = 8*half + j; j>=8 -> k = 16 + 8*half + (j-8)
    Frag a[2];
    #pragma unroll
    for (int mi = 0; mi < 2; ++mi) {
      const unsigned short* p = &As[cur][(wm * 32 + mi * 16 + l16) * 40 + half * 8];
      a[mi].u[0] = *(const u32x4*)p;
      a[mi].u[1] = *(const u32x4*)(p + 16);
    }
    // B fragment: lane col = n; halves j<8 -> k = 8*half + j; j>=8 -> k = 16 + 8*half + (j-8)
    #pragma unroll
    for (int nj = 0; nj < 4; ++nj) {
      Frag b;
      const unsigned short* p = &Bs[cur][(wn * 64 + nj * 16 + l16) * 40 + half * 8];
      b.u[0] = *(const u32x4*)p;
      b.u[1] = *(const u32x4*)(p + 16);
      acc[0][nj] = __builtin_amdgcn_wmma_f32_16x16x32_bf16(
          false, a[0].v, false, b.v, (short)0, acc[0][nj], false, false);
      acc[1][nj] = __builtin_amdgcn_wmma_f32_16x16x32_bf16(
          false, a[1].v, false, b.v, (short)0, acc[1][nj], false, false);
    }

    if (it + 1 < 32) storeTiles(1 - cur);
    __syncthreads();
  }

  // ---- epilogue: C/D layout: element i -> row = 8*half + i, col = l16 ----
  #pragma unroll
  for (int mi = 0; mi < 2; ++mi) {
    #pragma unroll
    for (int nj = 0; nj < 4; ++nj) {
      const int gn = nBlock + wn * 64 + nj * 16 + l16;
      const float bias_v = bias[gn];
      #pragma unroll
      for (int i = 0; i < 8; ++i) {
        const int gm = mBlock + wm * 32 + mi * 16 + half * 8 + i;
        const float v = acc[mi][nj][i] + bias_v;
        if (mode == 3) {
          out32[(size_t)gm * D_MODEL + gn] = v;
        } else {
          const int bb = gm >> 11, t = gm & (T_SEQ - 1);
          const int hh = gn >> 6,  hd = gn & (HEAD_DIM - 1);
          size_t idx;
          if (mode == 2)
            idx = ((size_t)(bb * NUM_HEADS + hh) * HEAD_DIM + hd) * T_SEQ + t;
          else
            idx = ((size_t)(bb * NUM_HEADS + hh) * T_SEQ + t) * HEAD_DIM + hd;
          out16[idx] = f32_to_bf16(v);
        }
      }
    }
  }
}

// ---------------------------------------------------------------------------
// Flash attention, transposed-score formulation, async-to-LDS double buffer.
//  Q,K : bf16 [B*H][T][64];  Vt : bf16 [B*H][64][T];  Ctx : bf16 [B][T][1024]
//  Each wave owns one 16-query tile; 8 waves/WG share LDS K/Vt tiles (same head).
//  K/Vt tiles staged with GLOBAL_LOAD_ASYNC_TO_LDS_B128 (ASYNCcnt, no VGPR pass).
//  S^T = K @ Q^T -> softmax stats per lane (+1 shfl_xor(16));
//  ctx^T += V^T @ P^T via WMMA with K-step 32.
// ---------------------------------------------------------------------------
__global__ __launch_bounds__(256) void flash_attn(
    const unsigned short* __restrict__ Q, const unsigned short* __restrict__ Kh,
    const unsigned short* __restrict__ Vt, unsigned short* __restrict__ Ctx)
{
  __shared__ unsigned short Ks[2][64 * 72];   // [key][d], stride 72 halves (144B)
  __shared__ unsigned short Vs[2][64 * 72];   // [d][key]

  const int tid  = threadIdx.x;
  const int wave = tid >> 5;
  const int lane = tid & 31;
  const int half = lane >> 4;
  const int l16  = lane & 15;
  const int bh   = blockIdx.x;                // 0..31
  const int b    = bh >> 4;
  const int h    = bh & 15;
  const int q0   = blockIdx.y * 128 + wave * 16;

  const size_t headQK = (size_t)bh * T_SEQ * HEAD_DIM;
  const size_t headV  = (size_t)bh * HEAD_DIM * T_SEQ;

  // async tile stage: each thread moves 32B of K and 32B of Vt (4 async b128 ops)
  const int sr = tid >> 2;                    // 0..63
  const int sc = (tid & 3) * 16;              // 0,16,32,48
  auto issueTile = [&](int buf, int kb) {
    const unsigned short* gk = Kh + headQK + (size_t)(kb + sr) * HEAD_DIM + sc;
    const unsigned short* gv = Vt + headV  + (size_t)sr * T_SEQ + kb + sc;
    unsigned short* lk = &Ks[buf][sr * 72 + sc];
    unsigned short* lv = &Vs[buf][sr * 72 + sc];
    __builtin_amdgcn_global_load_async_to_lds_b128(
        (GLOBAL_AS i32x4*)gk, (LDS_AS i32x4*)lk, 0, 0);
    __builtin_amdgcn_global_load_async_to_lds_b128(
        (GLOBAL_AS i32x4*)gk, (LDS_AS i32x4*)lk, 16, 0);
    __builtin_amdgcn_global_load_async_to_lds_b128(
        (GLOBAL_AS i32x4*)gv, (LDS_AS i32x4*)lv, 0, 0);
    __builtin_amdgcn_global_load_async_to_lds_b128(
        (GLOBAL_AS i32x4*)gv, (LDS_AS i32x4*)lv, 16, 0);
  };

  // Q^T B-fragments (col = query, k = head dim), loaded once
  v16bf qb[2];
  {
    const unsigned short* qp = Q + headQK + (size_t)(q0 + l16) * HEAD_DIM;
    #pragma unroll
    for (int dblk = 0; dblk < 2; ++dblk) {
      Frag f;
      const unsigned short* p = qp + dblk * 32 + half * 8;
      f.u[0] = *(const u32x4*)p;
      f.u[1] = *(const u32x4*)(p + 16);
      qb[dblk] = f.v;
    }
  }

  float m_run = -3.0e38f;
  float l_run = 0.0f;
  v8f o[4];                                   // ctx^T accumulators
  #pragma unroll
  for (int dr = 0; dr < 4; ++dr)
    #pragma unroll
    for (int i = 0; i < 8; ++i) o[dr][i] = 0.0f;

  const float scale = 0.125f;                 // 1/sqrt(64)

  issueTile(0, 0);

  for (int it = 0; it < T_SEQ / 64; ++it) {
    const int cur = it & 1;
    if (it + 1 < T_SEQ / 64) {
      issueTile(1 - cur, (it + 1) * 64);
      // async loads complete in order: <=4 outstanding drains this tile's 4 ops
      asm volatile("s_wait_asynccnt 0x4" ::: "memory");
    } else {
      asm volatile("s_wait_asynccnt 0x0" ::: "memory");
    }
    __syncthreads();                          // all waves' portions visible

    #pragma unroll
    for (int a2 = 0; a2 < 2; ++a2) {          // two 32-key blocks per LDS tile
      float s0[8], s1[8];
      #pragma unroll
      for (int t = 0; t < 2; ++t) {           // two 16-key S^T sub-tiles
        v8f s;
        #pragma unroll
        for (int i = 0; i < 8; ++i) s[i] = 0.0f;
        #pragma unroll
        for (int dblk = 0; dblk < 2; ++dblk) {
          Frag fa;                            // A = K rows (keys), k = head dims
          const unsigned short* p =
              &Ks[cur][(a2 * 32 + t * 16 + l16) * 72 + dblk * 32 + half * 8];
          fa.u[0] = *(const u32x4*)p;
          fa.u[1] = *(const u32x4*)(p + 16);
          s = __builtin_amdgcn_wmma_f32_16x16x32_bf16(
              false, fa.v, false, qb[dblk], (short)0, s, false, false);
        }
        float* dst = t ? s1 : s0;
        #pragma unroll
        for (int i = 0; i < 8; ++i) dst[i] = s[i] * scale;
      }

      // joint online softmax over these 32 keys (per query column)
      float tmax = s0[0];
      #pragma unroll
      for (int i = 1; i < 8; ++i) tmax = fmaxf(tmax, s0[i]);
      #pragma unroll
      for (int i = 0; i < 8; ++i) tmax = fmaxf(tmax, s1[i]);
      tmax = fmaxf(tmax, __shfl_xor(tmax, 16, 32));
      const float m_new = fmaxf(m_run, tmax);
      const float alpha = __expf(m_run - m_new);
      float p0[8], p1[8];
      float lsum = 0.0f;
      #pragma unroll
      for (int i = 0; i < 8; ++i) { p0[i] = __expf(s0[i] - m_new); lsum += p0[i]; }
      #pragma unroll
      for (int i = 0; i < 8; ++i) { p1[i] = __expf(s1[i] - m_new); lsum += p1[i]; }
      lsum += __shfl_xor(lsum, 16, 32);
      l_run = l_run * alpha + lsum;
      m_run = m_new;
      #pragma unroll
      for (int dr = 0; dr < 4; ++dr)
        #pragma unroll
        for (int i = 0; i < 8; ++i) o[dr][i] *= alpha;

      // P^T (D layout) -> B fragment (32 keys x 16 queries) via half-exchange
      union { unsigned short s[16]; v16bf v; } pb;
      #pragma unroll
      for (int i = 0; i < 8; ++i) {
        const float x0 = __shfl_xor(p0[i], 16, 32);
        const float x1 = __shfl_xor(p1[i], 16, 32);
        pb.s[i]     = f32_to_bf16(half ? x1 : p0[i]);
        pb.s[8 + i] = f32_to_bf16(half ? p1[i] : x0);
      }

      // ctx^T += V^T @ P^T  (A = Vs rows = dims, k = keys)
      #pragma unroll
      for (int dr = 0; dr < 4; ++dr) {
        Frag fa;
        const unsigned short* p =
            &Vs[cur][(dr * 16 + l16) * 72 + a2 * 32 + half * 16];
        fa.u[0] = *(const u32x4*)p;
        fa.u[1] = *(const u32x4*)(p + 8);
        o[dr] = __builtin_amdgcn_wmma_f32_16x16x32_bf16(
            false, fa.v, false, pb.v, (short)0, o[dr], false, false);
      }
    }
    __syncthreads();                          // done reading cur before overwrite
  }

  const float inv = 1.0f / l_run;
  const int q = q0 + l16;
  #pragma unroll
  for (int dr = 0; dr < 4; ++dr) {
    union { unsigned short s[8]; u32x4 u; } h8;
    #pragma unroll
    for (int i = 0; i < 8; ++i) h8.s[i] = f32_to_bf16(o[dr][i] * inv);
    const size_t base = ((size_t)(b * T_SEQ + q)) * D_MODEL
                      + h * HEAD_DIM + dr * 16 + half * 8;
    *(u32x4*)(Ctx + base) = h8.u;
  }
}

// ---------------------------------------------------------------------------
extern "C" void kernel_launch(void* const* d_in, const int* in_sizes, int n_in,
                              void* d_out, int out_size, void* d_ws, size_t ws_size,
                              hipStream_t stream) {
  (void)in_sizes; (void)n_in; (void)out_size; (void)ws_size;
  const float* x  = (const float*)d_in[0];
  const float* Wq = (const float*)d_in[1];
  const float* bq = (const float*)d_in[2];
  const float* Wk = (const float*)d_in[3];
  const float* bk = (const float*)d_in[4];
  const float* Wv = (const float*)d_in[5];
  const float* bv = (const float*)d_in[6];
  const float* Wo = (const float*)d_in[7];
  const float* bo = (const float*)d_in[8];
  float* out = (float*)d_out;

  const size_t HALF_CNT = (size_t)B_BATCH * NUM_HEADS * T_SEQ * HEAD_DIM; // 4 Mi halves
  unsigned short* Qbf  = (unsigned short*)d_ws;
  unsigned short* Kbf  = Qbf  + HALF_CNT;
  unsigned short* Vtbf = Kbf  + HALF_CNT;
  unsigned short* Ctx  = Vtbf + HALF_CNT;

  dim3 blk(256);
  dim3 ggrid(D_MODEL / 128, M_ROWS / 128);    // (8, 32)

  gemm_bf16wmma<<<ggrid, blk, 0, stream>>>(x, nullptr, Wq, bq, Qbf,  nullptr, 0);
  gemm_bf16wmma<<<ggrid, blk, 0, stream>>>(x, nullptr, Wk, bk, Kbf,  nullptr, 0);
  gemm_bf16wmma<<<ggrid, blk, 0, stream>>>(x, nullptr, Wv, bv, Vtbf, nullptr, 2);

  flash_attn<<<dim3(B_BATCH * NUM_HEADS, T_SEQ / 128), blk, 0, stream>>>(Qbf, Kbf, Vtbf, Ctx);

  gemm_bf16wmma<<<ggrid, blk, 0, stream>>>(nullptr, Ctx, Wo, bo, nullptr, out, 3);
}
